// NeighbourhoodTransformerBlock_18769007083807
// MI455X (gfx1250) — matrix operand order repaired and checked by reading, not compiled, not run on hardware
//
#include <hip/hip_runtime.h>
#include <hip/hip_bf16.h>

typedef __attribute__((ext_vector_type(16))) _Float16 v16h;
typedef __attribute__((ext_vector_type(8)))  _Float16 v8h;
typedef __attribute__((ext_vector_type(8)))  float    v8f;

#define N_TOK   25600     // b*t*H*W = 16*40*40
#define C_DIM   128
#define M_DIM   512
#define HH      40
#define WW      40
#define HEADS   4
#define DH      32
#define KNB     7

// ---- CDNA5 async global->LDS copy (16B per lane), tracked by ASYNCcnt ----
__device__ __forceinline__ void async_copy_b128(unsigned int lds_byte_off, const void* gaddr) {
    asm volatile("global_load_async_to_lds_b128 %0, %1, off"
                 :: "v"(lds_byte_off), "v"((unsigned long long)(uintptr_t)gaddr)
                 : "memory");
}

// ---------------- weight convert + transpose: w[K][Kout] f32 -> wt[Kout][K] f16 --------
__global__ void convert_transpose_kernel(const float* __restrict__ w, _Float16* __restrict__ wt,
                                         int K, int Kout) {
    size_t i = (size_t)blockIdx.x * 256 + threadIdx.x;      // over K*Kout, exact multiple
    int n = (int)(i % Kout);
    int k = (int)(i / Kout);
    wt[(size_t)n * K + k] = (_Float16)w[i];
}

// ---------------- input transpose (B,C,T,H,W) -> (N, C) ----------------
__global__ void transpose_in_kernel(const float* __restrict__ x, float* __restrict__ h) {
    size_t i = (size_t)blockIdx.x * 256 + threadIdx.x;   // over N_TOK*C_DIM
    int c  = (int)(i & 127);
    size_t n = i >> 7;
    int sp = (int)(n % 1600);
    int bt = (int)(n / 1600);
    int bb = bt >> 3, t = bt & 7;
    h[i] = x[(size_t)((bb * C_DIM + c) * 8 + t) * 1600 + sp];
}

// ---------------- LayerNorm (f32 in) -> f16 out ----------------
__global__ __launch_bounds__(256) void ln_half_kernel(
    const float* __restrict__ in, const float* __restrict__ g, const float* __restrict__ b,
    _Float16* __restrict__ outh)
{
    int wid = threadIdx.x >> 5, lane = threadIdx.x & 31;
    size_t row = (size_t)blockIdx.x * 8 + wid;
    const float* p = in + row * C_DIM;
    float v[4]; float s = 0.f, s2 = 0.f;
#pragma unroll
    for (int i = 0; i < 4; ++i) { v[i] = p[lane + 32 * i]; s += v[i]; s2 += v[i] * v[i]; }
#pragma unroll
    for (int m = 16; m >= 1; m >>= 1) { s += __shfl_xor(s, m, 32); s2 += __shfl_xor(s2, m, 32); }
    float mean = s * (1.f / 128.f);
    float var  = s2 * (1.f / 128.f) - mean * mean;
    float r    = rsqrtf(var + 1e-5f);
#pragma unroll
    for (int i = 0; i < 4; ++i) {
        int c = lane + 32 * i;
        outh[row * C_DIM + c] = (_Float16)((v[i] - mean) * r * g[c] + b[c]);
    }
}

// ---------------- final LayerNorm + transpose to (B,C,T,H,W) ----------------
__global__ __launch_bounds__(256) void ln_final_kernel(
    const float* __restrict__ in, const float* __restrict__ g, const float* __restrict__ b,
    float* __restrict__ out)
{
    int wid = threadIdx.x >> 5, lane = threadIdx.x & 31;
    size_t row = (size_t)blockIdx.x * 8 + wid;
    const float* p = in + row * C_DIM;
    float v[4]; float s = 0.f, s2 = 0.f;
#pragma unroll
    for (int i = 0; i < 4; ++i) { v[i] = p[lane + 32 * i]; s += v[i]; s2 += v[i] * v[i]; }
#pragma unroll
    for (int m = 16; m >= 1; m >>= 1) { s += __shfl_xor(s, m, 32); s2 += __shfl_xor(s2, m, 32); }
    float mean = s * (1.f / 128.f);
    float var  = s2 * (1.f / 128.f) - mean * mean;
    float r    = rsqrtf(var + 1e-5f);
    int sp = (int)(row % 1600);
    int bt = (int)(row / 1600);
    int bb = bt >> 3, t = bt & 7;
#pragma unroll
    for (int i = 0; i < 4; ++i) {
        int c = lane + 32 * i;
        float val = (v[i] - mean) * r * g[c] + b[c];
        out[(size_t)((bb * C_DIM + c) * 8 + t) * 1600 + sp] = val;
    }
}

// ---------------- WMMA GEMM with double-buffered async global->LDS tiles ----------------
// out[N x Kout] = f16(A[N x K]) * f16(Wt[Kout x K]^T) + bias (+res)
// block = 256 threads = 8 waves (4x2), tile 128x64, each wave 32x32 (2x2 wmma), BK=32.
template<bool RES>
__global__ __launch_bounds__(256) void gemm_wmma_kernel(
    const _Float16* __restrict__ A, const _Float16* __restrict__ Bt,
    const float* __restrict__ bias, const float* __restrict__ res,
    float* __restrict__ out, int K, int Kout)
{
    __shared__ _Float16 sA[2][128 * 32];   // [m][k]
    __shared__ _Float16 sB[2][64 * 32];    // [n][k]
    const int tid  = threadIdx.x;
    const int wid  = tid >> 5, lane = tid & 31;
    const int hi   = lane >> 4, mn = lane & 15;
    const int bm   = blockIdx.y * 128;
    const int bn   = blockIdx.x * 64;
    const int wm   = (wid & 3) * 32;
    const int wn   = (wid >> 2) * 32;

    // per-thread copy assignments (16B chunks)
    const int ra = tid >> 1, ca = (tid & 1) * 16;   // A: 128 rows x 32 halfs, 2 chunks/thread
    const int rb = tid >> 2, cb = (tid & 3) * 8;    // B: 64 rows x 32 halfs, 1 chunk/thread
    const _Float16* gA = A  + (size_t)(bm + ra) * K + ca;
    const _Float16* gB = Bt + (size_t)(bn + rb) * K + cb;
    unsigned int lA0[2], lA1[2], lB[2];
#pragma unroll
    for (int bf = 0; bf < 2; ++bf) {
        lA0[bf] = (unsigned int)(uintptr_t)&sA[bf][ra * 32 + ca];
        lA1[bf] = (unsigned int)(uintptr_t)&sA[bf][ra * 32 + ca + 8];
        lB[bf]  = (unsigned int)(uintptr_t)&sB[bf][rb * 32 + cb];
    }

    v8f zero = {0.f,0.f,0.f,0.f,0.f,0.f,0.f,0.f};
    v8f acc[2][2];
#pragma unroll
    for (int i = 0; i < 2; ++i)
#pragma unroll
        for (int j = 0; j < 2; ++j) acc[i][j] = zero;

    const int nk = K >> 5;
    // prologue: tile 0 -> buf 0 (3 async ops per thread)
    async_copy_b128(lA0[0], gA);
    async_copy_b128(lA1[0], gA + 8);
    async_copy_b128(lB[0],  gB);

    for (int ks = 0; ks < nk; ++ks) {
        const int buf = ks & 1;
        if (ks + 1 < nk) {
            const _Float16* a = gA + (ks + 1) * 32;
            const _Float16* b = gB + (ks + 1) * 32;
            async_copy_b128(lA0[buf ^ 1], a);
            async_copy_b128(lA1[buf ^ 1], a + 8);
            async_copy_b128(lB[buf ^ 1],  b);
            asm volatile("s_wait_asynccnt 0x3" ::: "memory");  // oldest 3 (current buf) done
        } else {
            asm volatile("s_wait_asynccnt 0x0" ::: "memory");
        }
        __syncthreads();

        v16h af[2], bfv[2];
#pragma unroll
        for (int s = 0; s < 2; ++s) {
            // A frag: m = mn; elems 0..7 -> k = hi*8 + e ; elems 8..15 -> k = 16 + hi*8 + e
            const v8h* a0 = (const v8h*)&sA[buf][(wm + s * 16 + mn) * 32 + hi * 8];
            const v8h* a1 = (const v8h*)&sA[buf][(wm + s * 16 + mn) * 32 + 16 + hi * 8];
#pragma unroll
            for (int e = 0; e < 8; ++e) { af[s][e] = (*a0)[e]; af[s][e + 8] = (*a1)[e]; }
            // B frag: n = mn; elems 0..7 -> k = hi*8 + e ; elems 8..15 -> k = 16 + hi*8 + e
            const v8h* b0 = (const v8h*)&sB[buf][(wn + s * 16 + mn) * 32 + hi * 8];
            const v8h* b1 = (const v8h*)&sB[buf][(wn + s * 16 + mn) * 32 + 16 + hi * 8];
#pragma unroll
            for (int e = 0; e < 8; ++e) { bfv[s][e] = (*b0)[e]; bfv[s][e + 8] = (*b1)[e]; }
        }
#pragma unroll
        for (int i = 0; i < 2; ++i)
#pragma unroll
            for (int j = 0; j < 2; ++j)
                acc[i][j] = __builtin_amdgcn_wmma_f32_16x16x32_f16(
                    false, af[i], false, bfv[j], (short)0, acc[i][j], false, false);
        __syncthreads();
    }

    // epilogue: C/D layout -> m = e + hi*8, n = mn
#pragma unroll
    for (int i = 0; i < 2; ++i)
#pragma unroll
        for (int j = 0; j < 2; ++j) {
            int col = bn + wn + j * 16 + mn;
            float bv = bias[col];
#pragma unroll
            for (int e = 0; e < 8; ++e) {
                int row = bm + wm + i * 16 + e + hi * 8;
                size_t o = (size_t)row * Kout + col;
                float v = acc[i][j][e] + bv;
                if (RES) v += res[o];
                out[o] = v;
            }
        }
}

// ---------------- 7x7 neighborhood attention, one wave per (token, head) -------------
__global__ __launch_bounds__(256) void na2d_attn_kernel(
    const float* __restrict__ qkv, const float* __restrict__ rpb,
    _Float16* __restrict__ outh)
{
    __shared__ float sl[8][64];
    int wid = threadIdx.x >> 5, lane = threadIdx.x & 31;
    int task = blockIdx.x * 8 + wid;             // N_TOK*HEADS tasks
    int hd = task & 3;
    int n  = task >> 2;
    int xx = n % WW, yy = (n / WW) % HH, bt = n / (HH * WW);
    int ys = yy - 3; ys = ys < 0 ? 0 : (ys > HH - KNB ? HH - KNB : ys);
    int xs = xx - 3; xs = xs < 0 ? 0 : (xs > WW - KNB ? WW - KNB : xs);
    const float scale = 0.17677669529663687f;    // 1/sqrt(32)
    float q = qkv[(size_t)n * 384 + hd * DH + lane] * scale;

    for (int j = 0; j < 49; ++j) {
        int a = j / 7, b = j % 7;
        int iy = ys + a, ix = xs + b;
        size_t nn = (size_t)((bt * HH + iy) * WW + ix);
        float p = q * qkv[nn * 384 + 128 + hd * DH + lane];
#pragma unroll
        for (int m = 16; m >= 1; m >>= 1) p += __shfl_xor(p, m, 32);
        if (lane == 0)
            sl[wid][j] = p + rpb[(hd * 13 + (iy - yy + 6)) * 13 + (ix - xx + 6)];
    }
    __syncthreads();

    float l0 = sl[wid][lane];
    float l1 = (lane < 17) ? sl[wid][32 + lane] : -1e30f;
    float mx = fmaxf(l0, l1);
#pragma unroll
    for (int m = 16; m >= 1; m >>= 1) mx = fmaxf(mx, __shfl_xor(mx, m, 32));
    float e0 = __expf(l0 - mx);
    float e1 = (lane < 17) ? __expf(l1 - mx) : 0.f;
    float s = e0 + e1;
#pragma unroll
    for (int m = 16; m >= 1; m >>= 1) s += __shfl_xor(s, m, 32);
    float inv = 1.f / s;
    __syncthreads();
    sl[wid][lane] = e0 * inv;
    if (lane < 17) sl[wid][32 + lane] = e1 * inv;
    __syncthreads();

    float o = 0.f;
    for (int j = 0; j < 49; ++j) {
        int a = j / 7, b = j % 7;
        int iy = ys + a, ix = xs + b;
        size_t nn = (size_t)((bt * HH + iy) * WW + ix);
        o += sl[wid][j] * qkv[nn * 384 + 256 + hd * DH + lane];
    }
    outh[(size_t)n * C_DIM + hd * DH + lane] = (_Float16)o;
}

// ---------------- depthwise 3x3x3 conv over (t,H,W) + exact GELU -> f16 --------------
__global__ __launch_bounds__(256) void dwconv_gelu_kernel(
    const float* __restrict__ in, const float* __restrict__ w, const float* __restrict__ bias,
    _Float16* __restrict__ outh)
{
    size_t i = (size_t)blockIdx.x * 256 + threadIdx.x;   // over N_TOK*M_DIM
    int m = (int)(i & (M_DIM - 1));
    size_t n = i >> 9;
    int xx = (int)(n % WW);
    int yy = (int)((n / WW) % HH);
    int bt = (int)(n / (HH * WW));
    int t = bt & 7, bb = bt >> 3;
    float acc = bias[m];
    for (int dt = -1; dt <= 1; ++dt) {
        int tt = t + dt; if (tt < 0 || tt > 7) continue;
        for (int dy = -1; dy <= 1; ++dy) {
            int iy = yy + dy; if (iy < 0 || iy > HH - 1) continue;
            for (int dx = -1; dx <= 1; ++dx) {
                int ix = xx + dx; if (ix < 0 || ix > WW - 1) continue;
                size_t nn = (size_t)(((bb * 8 + tt) * HH + iy) * WW + ix);
                acc += w[(((dt + 1) * 3 + (dy + 1)) * 3 + (dx + 1)) * M_DIM + m]
                       * in[nn * M_DIM + m];
            }
        }
    }
    float gel = 0.5f * acc * (1.f + erff(acc * 0.70710678118654752f));
    outh[i] = (_Float16)gel;
}

// ---------------- workspace layout (bytes) ----------------
#define WS_H    ((size_t)0)                        // f32 residual stream, N*128*4      = 13,107,200
#define WS_XH   ((size_t)13107200)                 // f16 activation buf,  N*512*2      = 26,214,400
#define WS_F1   ((size_t)39321600)                 // f32 scratch (qkv N*384 / fc1 N*512) = 52,428,800
#define WS_WH   ((size_t)91750400)                 // f16 transposed weights, 393216*2 bytes
// f16 weight offsets (elements); weights stored transposed [Kout][K]
#define WO_QKV  0          // L*384*128, layer stride 49152
#define WO_PROJ 98304      // L*128*128, layer stride 16384
#define WO_FC1  131072     // L*512*128, layer stride 65536
#define WO_FC2  262144     // L*128*512, layer stride 65536

extern "C" void kernel_launch(void* const* d_in, const int* in_sizes, int n_in,
                              void* d_out, int out_size, void* d_ws, size_t ws_size,
                              hipStream_t stream) {
    const float* x      = (const float*)d_in[0];
    const float* ln1_g  = (const float*)d_in[1];
    const float* ln1_b  = (const float*)d_in[2];
    const float* qkv_w  = (const float*)d_in[3];
    const float* qkv_b  = (const float*)d_in[4];
    const float* rpb    = (const float*)d_in[5];
    const float* proj_w = (const float*)d_in[6];
    const float* proj_b = (const float*)d_in[7];
    const float* ln2_g  = (const float*)d_in[8];
    const float* ln2_b  = (const float*)d_in[9];
    const float* fc1_w  = (const float*)d_in[10];
    const float* fc1_b  = (const float*)d_in[11];
    const float* dw_w   = (const float*)d_in[12];
    const float* dw_b   = (const float*)d_in[13];
    const float* fc2_w  = (const float*)d_in[14];
    const float* fc2_b  = (const float*)d_in[15];
    const float* out_g  = (const float*)d_in[16];
    const float* out_b  = (const float*)d_in[17];
    float* out = (float*)d_out;

    char* ws = (char*)d_ws;
    float*    h   = (float*)(ws + WS_H);
    _Float16* xh  = (_Float16*)(ws + WS_XH);
    float*    f1  = (float*)(ws + WS_F1);       // also qkv buffer (reused)
    float*    qkv = f1;
    _Float16* wh  = (_Float16*)(ws + WS_WH);

    // convert + transpose weights to f16 [Kout][K] (per layer)
    for (int i = 0; i < 2; ++i) {
        convert_transpose_kernel<<<(128*384)/256, 256, 0, stream>>>(
            qkv_w  + i*128*384, wh + WO_QKV  + i*49152, 128, 384);
        convert_transpose_kernel<<<(128*128)/256, 256, 0, stream>>>(
            proj_w + i*128*128, wh + WO_PROJ + i*16384, 128, 128);
        convert_transpose_kernel<<<(128*512)/256, 256, 0, stream>>>(
            fc1_w  + i*128*512, wh + WO_FC1  + i*65536, 128, 512);
        convert_transpose_kernel<<<(512*128)/256, 256, 0, stream>>>(
            fc2_w  + i*512*128, wh + WO_FC2  + i*65536, 512, 128);
    }

    // (B,C,T,H,W) -> (N,C)
    transpose_in_kernel<<<(N_TOK*C_DIM)/256, 256, 0, stream>>>(x, h);

    for (int i = 0; i < 2; ++i) {
        // --- attention half ---
        ln_half_kernel<<<N_TOK/8, 256, 0, stream>>>(h, ln1_g + i*C_DIM, ln1_b + i*C_DIM, xh);
        gemm_wmma_kernel<false><<<dim3(384/64, N_TOK/128), 256, 0, stream>>>(
            xh, wh + WO_QKV + i*49152, qkv_b + i*384, nullptr, qkv, 128, 384);
        na2d_attn_kernel<<<(N_TOK*HEADS)/8, 256, 0, stream>>>(qkv, rpb + i*HEADS*13*13, xh);
        gemm_wmma_kernel<true><<<dim3(128/64, N_TOK/128), 256, 0, stream>>>(
            xh, wh + WO_PROJ + i*16384, proj_b + i*C_DIM, h, h, 128, 128);
        // --- FFN half ---
        ln_half_kernel<<<N_TOK/8, 256, 0, stream>>>(h, ln2_g + i*C_DIM, ln2_b + i*C_DIM, xh);
        gemm_wmma_kernel<false><<<dim3(512/64, N_TOK/128), 256, 0, stream>>>(
            xh, wh + WO_FC1 + i*65536, fc1_b + i*M_DIM, nullptr, f1, 128, 512);
        dwconv_gelu_kernel<<<(N_TOK*M_DIM)/256, 256, 0, stream>>>(
            f1, dw_w + i*27*M_DIM, dw_b + i*M_DIM, xh);
        gemm_wmma_kernel<true><<<dim3(128/64, N_TOK/128), 256, 0, stream>>>(
            xh, wh + WO_FC2 + i*65536, fc2_b + i*C_DIM, h, h, 512, 128);
    }

    ln_final_kernel<<<N_TOK/8, 256, 0, stream>>>(h, out_g, out_b, out);
}
